// Attention_5471788335537
// MI455X (gfx1250) — matrix operand order repaired and checked by reading, not compiled, not run on hardware
//
#include <hip/hip_runtime.h>
#include <math.h>

// ---------------- problem constants ----------------
#define BATCH   8
#define DIM     256          // input channels
#define NPIX    1024         // 32*32
#define HEADS   8
#define DHEAD   64
#define INNER   512          // HEADS*DHEAD
#define SCALE_F 0.125f       // 64^-0.5

typedef __attribute__((ext_vector_type(16))) __bf16 v16bf;
typedef __attribute__((ext_vector_type(8)))  __bf16 v8bf;
typedef __attribute__((ext_vector_type(8)))  float  v8f;

static __device__ __forceinline__ v8f wmma_bf16(v16bf a, v16bf b, v8f c) {
    return __builtin_amdgcn_wmma_f32_16x16x32_bf16(false, a, false, b, (short)0, c, false, false);
}

// A fragment (lane owns row m, K-mapping: k=kbase+e)
// 'kbase' must already include k0 + 16*hh. Two 16-byte contiguous loads.
static __device__ __forceinline__ v16bf load_afrag(const __bf16* __restrict__ row, int kbase) {
    v8bf lo = *(const v8bf*)(row + kbase);
    v8bf hi = *(const v8bf*)(row + kbase + 8);
    return __builtin_shufflevector(lo, hi, 0,1,2,3,4,5,6,7,8,9,10,11,12,13,14,15);
}
// B fragment (lane owns col n, k = kbase + e, kbase includes k0 + 16*hh): one 32B chunk.
static __device__ __forceinline__ v16bf load_bfrag(const __bf16* __restrict__ p) {
    v8bf lo = *(const v8bf*)(p);
    v8bf hi = *(const v8bf*)(p + 8);
    return __builtin_shufflevector(lo, hi, 0,1,2,3,4,5,6,7,8,9,10,11,12,13,14,15);
}

// ---------------- conversions / transposes ----------------
// Xt[(b*1024+pix)*256 + c] = x[(b*256+c)*1024 + pix]
__global__ void cvt_x_t(const float* __restrict__ src, __bf16* __restrict__ dst) {
    int total = BATCH * DIM * NPIX;
    for (int i = blockIdx.x * blockDim.x + threadIdx.x; i < total; i += gridDim.x * blockDim.x) {
        int c   = i & (DIM - 1);
        int pix = (i >> 8) & (NPIX - 1);
        int b   = i >> 18;
        dst[i] = (__bf16)src[(((b << 8) + c) << 10) + pix];
    }
}
// Wkvt[((t*1024 + m)*256) + c] = Wkv[(m*256 + c)*9 + t]
__global__ void cvt_wkv_t(const float* __restrict__ src, __bf16* __restrict__ dst) {
    int total = 9 * 1024 * DIM;
    for (int i = blockIdx.x * blockDim.x + threadIdx.x; i < total; i += gridDim.x * blockDim.x) {
        int c = i & (DIM - 1);
        int m = (i >> 8) & 1023;
        int t = i >> 18;
        dst[i] = (__bf16)src[(((m << 8) + c) * 9) + t];
    }
}
__global__ void cvt_plain(const float* __restrict__ src, __bf16* __restrict__ dst, int n) {
    for (int i = blockIdx.x * blockDim.x + threadIdx.x; i < n; i += gridDim.x * blockDim.x)
        dst[i] = (__bf16)src[i];
}

// ---------------- q: M=1024 pixels, N=512 cout, K=256 ----------------
__global__ void gemm_q(const __bf16* __restrict__ Wq, const __bf16* __restrict__ Xt,
                       __bf16* __restrict__ Qt) {
    int wave = blockIdx.x * (blockDim.x >> 5) + (threadIdx.x >> 5);
    int lane = threadIdx.x & 31, lm = lane & 15, hh = lane >> 4;
    int nt = wave & 15;           // 16 cout tiles of 32
    int mt = (wave >> 4) & 31;    // 32 pixel tiles of 32
    int b  = wave >> 9;
    int m0 = mt * 32, n0 = nt * 32;

    const __bf16* xrow0 = Xt + ((b << 10) + m0 + lm) * DIM;
    const __bf16* xrow1 = xrow0 + 16 * DIM;
    const __bf16* w0 = Wq + (n0 + lm) * DIM;
    const __bf16* w1 = w0 + 16 * DIM;

    v8f a00 = {}, a01 = {}, a10 = {}, a11 = {};
    for (int k0 = 0; k0 < DIM; k0 += 32) {
        int ka = k0 + 16 * hh, kb = k0 + 16 * hh;
        v16bf fa0 = load_afrag(xrow0, ka);
        v16bf fa1 = load_afrag(xrow1, ka);
        v16bf fb0 = load_bfrag(w0 + kb);
        v16bf fb1 = load_bfrag(w1 + kb);
        a00 = wmma_bf16(fa0, fb0, a00); a01 = wmma_bf16(fa0, fb1, a01);
        a10 = wmma_bf16(fa1, fb0, a10); a11 = wmma_bf16(fa1, fb1, a11);
    }
    __bf16* q = Qt + (size_t)(b << 10) * INNER;
#pragma unroll
    for (int r = 0; r < 8; ++r) {
        int pA = m0 + r + 8 * hh, pB = pA + 16;
        q[pA * INNER + n0 + lm]      = (__bf16)a00[r];
        q[pA * INNER + n0 + 16 + lm] = (__bf16)a01[r];
        q[pB * INNER + n0 + lm]      = (__bf16)a10[r];
        q[pB * INNER + n0 + 16 + lm] = (__bf16)a11[r];
    }
}

// ---------------- kv conv3x3: M=1024 pixels, N=1024 cout, K=9*256 ----------------
__global__ void conv_kv(const __bf16* __restrict__ Wkvt, const __bf16* __restrict__ Xt,
                        __bf16* __restrict__ Kt, __bf16* __restrict__ Vcm) {
    int wave = blockIdx.x * (blockDim.x >> 5) + (threadIdx.x >> 5);
    int lane = threadIdx.x & 31, lm = lane & 15, hh = lane >> 4;
    int nt = wave & 31;           // 32 cout tiles of 32
    int mt = (wave >> 5) & 31;    // 32 pixel tiles of 32
    int b  = wave >> 10;
    int m0 = mt * 32, n0 = nt * 32;

    int pA = m0 + lm, pB = pA + 16;
    int yA = pA >> 5, xA = pA & 31;
    int yB = pB >> 5, xB = pB & 31;
    const __bf16* xbase = Xt + (size_t)(b << 10) * DIM;

    v8f a00 = {}, a01 = {}, a10 = {}, a11 = {};
    for (int t = 0; t < 9; ++t) {
        int dy = t / 3 - 1, dx = t % 3 - 1;
        int pyA = yA + dy, pxA = xA + dx;
        int pyB = yB + dy, pxB = xB + dx;
        bool vA = ((unsigned)pyA < 32u) && ((unsigned)pxA < 32u);
        bool vB = ((unsigned)pyB < 32u) && ((unsigned)pxB < 32u);
        const __bf16* rowA = xbase + (vA ? (pyA * 32 + pxA) * DIM : 0);
        const __bf16* rowB = xbase + (vB ? (pyB * 32 + pxB) * DIM : 0);
        const __bf16* w0 = Wkvt + ((t << 10) + n0 + lm) * DIM;
        const __bf16* w1 = w0 + 16 * DIM;
        for (int k0 = 0; k0 < DIM; k0 += 32) {
            int ka = k0 + 16 * hh, kb = k0 + 16 * hh;
            v16bf fa0 = load_afrag(rowA, ka);
            v16bf fa1 = load_afrag(rowB, ka);
            if (!vA) fa0 = (v16bf)(__bf16)0.0f;
            if (!vB) fa1 = (v16bf)(__bf16)0.0f;
            v16bf fb0 = load_bfrag(w0 + kb);
            v16bf fb1 = load_bfrag(w1 + kb);
            a00 = wmma_bf16(fa0, fb0, a00); a01 = wmma_bf16(fa0, fb1, a01);
            a10 = wmma_bf16(fa1, fb0, a10); a11 = wmma_bf16(fa1, fb1, a11);
        }
    }
    bool isK = (n0 < INNER);    // uniform per wave
#pragma unroll
    for (int r = 0; r < 8; ++r) {
        int rowA = m0 + r + 8 * hh, rowB = rowA + 16;
        int c0 = n0 + lm, c1 = n0 + 16 + lm;
        if (isK) {
            __bf16* k = Kt + (size_t)(b << 10) * INNER;
            k[rowA * INNER + c0] = (__bf16)a00[r];
            k[rowA * INNER + c1] = (__bf16)a01[r];
            k[rowB * INNER + c0] = (__bf16)a10[r];
            k[rowB * INNER + c1] = (__bf16)a11[r];
        } else {
            __bf16* v = Vcm + (size_t)b * INNER * NPIX;
            v[(c0 - INNER) * NPIX + rowA] = (__bf16)a00[r];
            v[(c1 - INNER) * NPIX + rowA] = (__bf16)a01[r];
            v[(c0 - INNER) * NPIX + rowB] = (__bf16)a10[r];
            v[(c1 - INNER) * NPIX + rowB] = (__bf16)a11[r];
        }
    }
}

// ---------------- flash attention ----------------
// Block = 4 waves, one (b,h), 128 query rows (32 per wave as 2 row-tiles).
// Per 32-key step the block stages K (as [key][d]) and V (as [d][key]) in LDS once.
__global__ void attn(const __bf16* __restrict__ Qt, const __bf16* __restrict__ Kt,
                     const __bf16* __restrict__ Vcm, __bf16* __restrict__ At) {
    __shared__ __align__(16) __bf16 lds_k[32 * 64];        // [key][d]   4 KB
    __shared__ __align__(16) __bf16 lds_v[64 * 32];        // [d][key]   4 KB
    __shared__ __align__(16) __bf16 lds_p[4][16 * 32];     // per-wave P 4 KB

    int tid = threadIdx.x;
    int wslot = tid >> 5;
    int lane = tid & 31, lm = lane & 15, hh = lane >> 4;
    int rb = blockIdx.x & 7;           // 8 row-blocks of 128 rows
    int bh = blockIdx.x >> 3;          // 64 (b,h) pairs
    int b = bh >> 3, h = bh & 7;
    int m0w = rb * 128 + wslot * 32;   // first of this wave's 32 rows

    const __bf16* qbase = Qt + (size_t)(b << 10) * INNER + h * DHEAD;
    const __bf16* kbase = Kt + (size_t)(b << 10) * INNER + h * DHEAD;
    const __bf16* vbase = Vcm + ((size_t)b * INNER + h * DHEAD) * NPIX;

    // Q fragments for both row tiles, resident all loop long
    v16bf qa[2][2];
#pragma unroll
    for (int rt = 0; rt < 2; ++rt)
#pragma unroll
        for (int kk = 0; kk < 2; ++kk)
            qa[rt][kk] = load_afrag(qbase + (m0w + rt * 16 + lm) * INNER, kk * 32 + 16 * hh);

    float mstate[2][8], lstate[2][8];
    v8f oacc[2][4];
#pragma unroll
    for (int rt = 0; rt < 2; ++rt) {
#pragma unroll
        for (int r = 0; r < 8; ++r) { mstate[rt][r] = -1e30f; lstate[rt][r] = 0.0f; }
#pragma unroll
        for (int dt = 0; dt < 4; ++dt) oacc[rt][dt] = (v8f){};
    }

    for (int j0 = 0; j0 < NPIX; j0 += 32) {
        __syncthreads();   // previous iteration's K/V readers done
        // ---- cooperative staging: 256 x 16B chunks each for K and V ----
#pragma unroll
        for (int cc = 0; cc < 2; ++cc) {
            int chunk = tid + cc * 128;
            // K: key = chunk/8, dchunk = chunk%8  (row = 128B contiguous)
            int key = chunk >> 3, dc = chunk & 7;
            *(v8bf*)&lds_k[key * 64 + dc * 8] =
                *(const v8bf*)(kbase + (j0 + key) * INNER + dc * 8);
            // V: d = chunk/4, kc = chunk%4 (row = 64B contiguous in pixels)
            int d = chunk >> 2, kc = chunk & 3;
            *(v8bf*)&lds_v[d * 32 + kc * 8] =
                *(const v8bf*)(vbase + (size_t)d * NPIX + j0 + kc * 8);
        }
        __syncthreads();   // staged tiles visible

        __bf16* pl = lds_p[wslot];
#pragma unroll
        for (int rt = 0; rt < 2; ++rt) {
            // ---- scores: two 16x16 C tiles over staged K ----
            v8f s0 = (v8f){}, s1 = (v8f){};
#pragma unroll
            for (int kk = 0; kk < 2; ++kk) {
                v16bf kf0 = load_bfrag(&lds_k[(lm) * 64 + kk * 32 + 16 * hh]);
                v16bf kf1 = load_bfrag(&lds_k[(16 + lm) * 64 + kk * 32 + 16 * hh]);
                s0 = wmma_bf16(qa[rt][kk], kf0, s0);
                s1 = wmma_bf16(qa[rt][kk], kf1, s1);
            }
            // ---- online softmax ----
#pragma unroll
            for (int r = 0; r < 8; ++r) {
                float v0 = s0[r] * SCALE_F;
                float v1 = s1[r] * SCALE_F;
                float t = fmaxf(v0, v1);
#pragma unroll
                for (int off = 1; off < 16; off <<= 1)
                    t = fmaxf(t, __shfl_xor(t, off, 16));
                float mnew  = fmaxf(mstate[rt][r], t);
                float alpha = __expf(mstate[rt][r] - mnew);
                float p0 = __expf(v0 - mnew);
                float p1 = __expf(v1 - mnew);
                float rs = p0 + p1;
#pragma unroll
                for (int off = 1; off < 16; off <<= 1)
                    rs += __shfl_xor(rs, off, 16);
                lstate[rt][r] = lstate[rt][r] * alpha + rs;
                mstate[rt][r] = mnew;
#pragma unroll
                for (int dt = 0; dt < 4; ++dt) oacc[rt][dt][r] *= alpha;
                int row = r + 8 * hh;
                pl[row * 32 + lm]      = (__bf16)p0;
                pl[row * 32 + 16 + lm] = (__bf16)p1;
            }
            // wave-internal RAW on pl: same-wave LDS ops are ordered; no barrier needed
            v16bf pa = load_afrag(pl + lm * 32, 16 * hh);
            // ---- O += P * V over staged V ----
#pragma unroll
            for (int dt = 0; dt < 4; ++dt) {
                v16bf vf = load_bfrag(&lds_v[(dt * 16 + lm) * 32 + 16 * hh]);
                oacc[rt][dt] = wmma_bf16(pa, vf, oacc[rt][dt]);
            }
        }
    }

    // ---- epilogue: normalize + exact GELU, store pixel-major At[b][pix][c] ----
#pragma unroll
    for (int rt = 0; rt < 2; ++rt) {
#pragma unroll
        for (int dt = 0; dt < 4; ++dt) {
#pragma unroll
            for (int r = 0; r < 8; ++r) {
                float val = oacc[rt][dt][r] / lstate[rt][r];
                float g = 0.5f * val * (1.0f + erff(val * 0.70710678118f));
                int row = m0w + rt * 16 + r + 8 * hh;
                At[((size_t)(b << 10) + row) * INNER + h * DHEAD + dt * 16 + lm] = (__bf16)g;
            }
        }
    }
}

// ---------------- out: M=256 cout, N=1024 pixels, K=512 ----------------
__global__ void gemm_out(const __bf16* __restrict__ Wout, const __bf16* __restrict__ At,
                         const float* __restrict__ bout, float* __restrict__ out) {
    int wave = blockIdx.x * (blockDim.x >> 5) + (threadIdx.x >> 5);
    int lane = threadIdx.x & 31, lm = lane & 15, hh = lane >> 4;
    int nt = wave & 31;           // 32 pixel tiles of 32
    int mt = (wave >> 5) & 7;     // 8 cout tiles of 32
    int b  = wave >> 8;
    int m0 = mt * 32, n0 = nt * 32;

    const __bf16* w0 = Wout + (m0 + lm) * INNER;
    const __bf16* w1 = w0 + 16 * INNER;
    const __bf16* ar0 = At + ((size_t)(b << 10) + n0 + lm) * INNER;
    const __bf16* ar1 = ar0 + 16 * INNER;

    v8f a00 = {}, a01 = {}, a10 = {}, a11 = {};
    for (int k0 = 0; k0 < INNER; k0 += 32) {
        int ka = k0 + 16 * hh, kb = k0 + 16 * hh;
        v16bf fa0 = load_afrag(w0, ka);
        v16bf fa1 = load_afrag(w1, ka);
        v16bf fb0 = load_bfrag(ar0 + kb);
        v16bf fb1 = load_bfrag(ar1 + kb);
        a00 = wmma_bf16(fa0, fb0, a00); a01 = wmma_bf16(fa0, fb1, a01);
        a10 = wmma_bf16(fa1, fb0, a10); a11 = wmma_bf16(fa1, fb1, a11);
    }
#pragma unroll
    for (int r = 0; r < 8; ++r) {
        int mA = m0 + r + 8 * hh, mB = mA + 16;
        int p0 = n0 + lm, p1 = n0 + 16 + lm;
        float* ob = out + (size_t)b * DIM * NPIX;
        ob[mA * NPIX + p0] = a00[r] + bout[mA];
        ob[mA * NPIX + p1] = a01[r] + bout[mA];
        ob[mB * NPIX + p0] = a10[r] + bout[mB];
        ob[mB * NPIX + p1] = a11[r] + bout[mB];
    }
}

// ---------------- launcher ----------------
extern "C" void kernel_launch(void* const* d_in, const int* in_sizes, int n_in,
                              void* d_out, int out_size, void* d_ws, size_t ws_size,
                              hipStream_t stream) {
    const float* x    = (const float*)d_in[0];
    const float* Wq   = (const float*)d_in[1];
    const float* Wkv  = (const float*)d_in[2];
    const float* Wout = (const float*)d_in[3];
    const float* bout = (const float*)d_in[4];
    float* out = (float*)d_out;

    const size_t nx    = (size_t)BATCH * DIM * NPIX;        // 2,097,152
    const size_t nwq   = (size_t)INNER * DIM;               // 131,072
    const size_t nwkv  = (size_t)2 * INNER * DIM * 9;       // 2,359,296
    const size_t nwout = (size_t)DIM * INNER;               // 131,072
    const size_t nq    = (size_t)BATCH * INNER * NPIX;      // 4,194,304

    __bf16* xt    = (__bf16*)d_ws;
    __bf16* wqb   = xt + nx;
    __bf16* wkvt  = wqb + nwq;
    __bf16* woutb = wkvt + nwkv;
    __bf16* qt    = woutb + nwout;
    __bf16* kt    = qt + nq;
    __bf16* vcm   = kt + nq;
    __bf16* abuf  = vcm + nq;

    cvt_x_t<<<2048, 256, 0, stream>>>(x, xt);
    cvt_plain<<<512, 256, 0, stream>>>(Wq, wqb, (int)nwq);
    cvt_wkv_t<<<2048, 256, 0, stream>>>(Wkv, wkvt);
    cvt_plain<<<512, 256, 0, stream>>>(Wout, woutb, (int)nwout);

    // gemm_q: 8 * 32 * 16 = 4096 waves -> 1024 blocks x 4 waves
    gemm_q<<<1024, 128, 0, stream>>>(wqb, xt, qt);
    // conv_kv: 8 * 32 * 32 = 8192 waves -> 2048 blocks
    conv_kv<<<2048, 128, 0, stream>>>(wkvt, xt, kt, vcm);
    // attn: 64 (b,h) * 8 row-blocks = 512 blocks x 128 threads
    attn<<<512, 128, 0, stream>>>(qt, kt, vcm, abuf);
    // gemm_out: 8 * 8 * 32 = 2048 waves -> 512 blocks
    gemm_out<<<512, 128, 0, stream>>>(woutb, abuf, bout, out);
}
